// LinearTransformerEncoderLayer_24404004176366
// MI455X (gfx1250) — matrix operand order repaired and checked by reading, not compiled, not run on hardware
//
#include <hip/hip_runtime.h>

// ---------------------------------------------------------------------------
// LinearTransformerEncoderLayer for MI455X (gfx1250, wave32, WMMA bf16).
// Compute-bound on the 4 big GEMMs (~825 GFLOP) -> everything runs through
// v_wmma_f32_16x16x32_bf16 with fp32 accumulation; pointwise ops fused into
// GEMM epilogues.  GEMM main loop is double-buffered in LDS; tiles are
// fetched with gfx1250 async global->LDS (ASYNCcnt) instructions.
// ---------------------------------------------------------------------------

typedef __bf16 v16bf __attribute__((ext_vector_type(16)));
typedef float  v8f   __attribute__((ext_vector_type(8)));
typedef int    v4i   __attribute__((vector_size(16)));   // 128-bit async payload

union FragU { uint4 u[2]; v16bf v; };   // 32 bytes: one WMMA A/B operand

static constexpr int Bb = 8, Nn = 4096, Cc = 1024, Hh = 16, Dd = 64, FFd = 4096;
static constexpr int Mrows = Bb * Nn;   // 32768 token rows

#if defined(__gfx1250__) && \
    __has_builtin(__builtin_amdgcn_global_load_async_to_lds_b128) && \
    __has_builtin(__builtin_amdgcn_s_wait_asynccnt)
#define USE_ASYNC_LDS 1
typedef __attribute__((address_space(1))) v4i* gv4i_p;   // global 128b
typedef __attribute__((address_space(3))) v4i* lv4i_p;   // LDS 128b
#else
#define USE_ASYNC_LDS 0
#endif

__device__ __forceinline__ unsigned short f2bf(float f) {
  unsigned u = __builtin_bit_cast(unsigned, f);
  u += 0x7fffu + ((u >> 16) & 1u);            // round-to-nearest-even
  return (unsigned short)(u >> 16);
}
__device__ __forceinline__ float bf2f(unsigned short h) {
  unsigned u = ((unsigned)h) << 16;
  return __builtin_bit_cast(float, u);
}

// ---------------------------------------------------------------------------
// fp32 -> bf16 elementwise (weights)
// ---------------------------------------------------------------------------
__global__ void f32_to_bf16_kernel(const float* __restrict__ in,
                                   unsigned short* __restrict__ out, int n) {
  int i = blockIdx.x * 256 + threadIdx.x;
  int stride = gridDim.x * 256;
  for (; i < n; i += stride) out[i] = f2bf(in[i]);
}

// ---------------------------------------------------------------------------
// LayerNorm over C=1024, one block per row, bf16 output
// ---------------------------------------------------------------------------
__global__ __launch_bounds__(256)
void layernorm_bf16_kernel(const float* __restrict__ x,
                           const float* __restrict__ g,
                           const float* __restrict__ bta,
                           unsigned short* __restrict__ out) {
  __shared__ float s1[256], s2[256];
  const int row = blockIdx.x;
  const int tid = threadIdx.x;
  const float* xr = x + (size_t)row * Cc;
  float lv[4];
  float sum = 0.f, sq = 0.f;
#pragma unroll
  for (int i = 0; i < 4; ++i) {
    float t = xr[tid + i * 256];
    lv[i] = t; sum += t; sq += t * t;
  }
  s1[tid] = sum; s2[tid] = sq;
  __syncthreads();
  for (int off = 128; off > 0; off >>= 1) {
    if (tid < off) { s1[tid] += s1[tid + off]; s2[tid] += s2[tid + off]; }
    __syncthreads();
  }
  const float mu  = s1[0] * (1.0f / Cc);
  const float var = s2[0] * (1.0f / Cc) - mu * mu;
  const float inv = rsqrtf(var + 1e-5f);
#pragma unroll
  for (int i = 0; i < 4; ++i) {
    int c = tid + i * 256;
    out[(size_t)row * Cc + c] = f2bf((lv[i] - mu) * inv * g[c] + bta[c]);
  }
}

// ---------------------------------------------------------------------------
// Tiled bf16 WMMA GEMM:  out = A[M,K] @ W[N,K]^T  (+ fused epilogue)
// Block tile 128x128, K staged 64 deep, double-buffered LDS, 8 waves of 64x32.
// ---------------------------------------------------------------------------
enum { EPI_RES_F32 = 0, EPI_GELU_BF16 = 1, EPI_QKV = 2 };

template <int EPI>
__global__ __launch_bounds__(256)
void gemm_bf16_kernel(const unsigned short* __restrict__ A,   // [M,K] bf16
                      const unsigned short* __restrict__ W,   // [N,K] bf16
                      const float* __restrict__ bias,         // [N]
                      const float* __restrict__ res,          // [M,N] or null
                      float* __restrict__ outF,
                      unsigned short* __restrict__ outH,
                      unsigned short* __restrict__ qf,        // [B,H,N,D]
                      unsigned short* __restrict__ kf,        // [B,H,D,N]
                      unsigned short* __restrict__ vv,        // [B,H,D,N]
                      const unsigned char* __restrict__ mask, // [B,N], 1 = PAD
                      int M, int N, int K) {
  constexpr int LDT = 72;  // LDS row stride (bf16), 144B = 9x16B, conflict pad
  __shared__ __align__(16) unsigned short As[2][128 * LDT];
  __shared__ __align__(16) unsigned short Bs[2][128 * LDT];

  const int tid  = threadIdx.x;
  const int lane = tid & 31, wave = tid >> 5;
  const int half = lane >> 4, mr = lane & 15;
  const int wr = wave >> 2, wc = wave & 3;      // 2x4 wave grid
  const int bm0 = blockIdx.y * 128;
  const int bn0 = blockIdx.x * 128;

  const v8f zero8 = {0.f, 0.f, 0.f, 0.f, 0.f, 0.f, 0.f, 0.f};
  v8f acc[4][2];
#pragma unroll
  for (int i = 0; i < 4; ++i)
#pragma unroll
    for (int j = 0; j < 2; ++j) acc[i][j] = zero8;

#if USE_ASYNC_LDS
  // async global->LDS tile fetch (8 x 16B per thread), tracked on ASYNCcnt
  auto issue_tile = [&](int k0, int buf) {
#pragma unroll
    for (int it = 0; it < 4; ++it) {
      int cid = tid + it * 256;
      int row = cid >> 3, cc = cid & 7;
      const unsigned short* ga = A + (size_t)(bm0 + row) * K + k0 + cc * 8;
      const unsigned short* gb = W + (size_t)(bn0 + row) * K + k0 + cc * 8;
      unsigned short* la = &As[buf][row * LDT + cc * 8];
      unsigned short* lb = &Bs[buf][row * LDT + cc * 8];
      __builtin_amdgcn_global_load_async_to_lds_b128((gv4i_p)ga, (lv4i_p)la, 0, 0);
      __builtin_amdgcn_global_load_async_to_lds_b128((gv4i_p)gb, (lv4i_p)lb, 0, 0);
    }
  };
  issue_tile(0, 0);
  __builtin_amdgcn_s_wait_asynccnt(0);
  __syncthreads();
#else
  // synchronous fill of buffer 0
#pragma unroll
  for (int it = 0; it < 4; ++it) {
    int cid = tid + it * 256;
    int row = cid >> 3, cc = cid & 7;
    *(uint4*)(&As[0][row * LDT + cc * 8]) =
        *(const uint4*)(A + (size_t)(bm0 + row) * K + cc * 8);
    *(uint4*)(&Bs[0][row * LDT + cc * 8]) =
        *(const uint4*)(W + (size_t)(bn0 + row) * K + cc * 8);
  }
  __syncthreads();
#endif

  int cur = 0;
  for (int k0 = 0; k0 < K; k0 += 64) {
    const bool hasNext = (k0 + 64) < K;

#if USE_ASYNC_LDS
    if (hasNext) issue_tile(k0 + 64, cur ^ 1);
#else
    uint4 pa[4], pb[4];
    if (hasNext) {
#pragma unroll
      for (int it = 0; it < 4; ++it) {
        int cid = tid + it * 256;
        int row = cid >> 3, cc = cid & 7;
        pa[it] = *(const uint4*)(A + (size_t)(bm0 + row) * K + (k0 + 64) + cc * 8);
        pb[it] = *(const uint4*)(W + (size_t)(bn0 + row) * K + (k0 + 64) + cc * 8);
      }
    }
#endif

    const unsigned short* Ab = &As[cur][0];
    const unsigned short* Bbse = &Bs[cur][0];
#pragma unroll
    for (int kk = 0; kk < 64; kk += 32) {
      FragU a[4], bq[2];
#pragma unroll
      for (int i = 0; i < 4; ++i) {   // A: row per lane, K interleaved by half
        const unsigned short* p = Ab + (wr * 64 + i * 16 + mr) * LDT + kk + 8 * half;
        a[i].u[0] = *(const uint4*)p;
        a[i].u[1] = *(const uint4*)(p + 16);
      }
#pragma unroll
      for (int j = 0; j < 2; ++j) {   // B: col per lane, K interleaved by half
        const unsigned short* p = Bbse + (wc * 32 + j * 16 + mr) * LDT + kk + 8 * half;
        bq[j].u[0] = *(const uint4*)p;
        bq[j].u[1] = *(const uint4*)(p + 16);
      }
#pragma unroll
      for (int i = 0; i < 4; ++i)
#pragma unroll
        for (int j = 0; j < 2; ++j)
          acc[i][j] = __builtin_amdgcn_wmma_f32_16x16x32_bf16(
              false, a[i].v, false, bq[j].v, (short)0, acc[i][j], false, false);
    }

#if USE_ASYNC_LDS
    if (hasNext) {
      __builtin_amdgcn_s_wait_asynccnt(0);
      __syncthreads();
    }
#else
    if (hasNext) {
#pragma unroll
      for (int it = 0; it < 4; ++it) {
        int cid = tid + it * 256;
        int row = cid >> 3, cc = cid & 7;
        *(uint4*)(&As[cur ^ 1][row * LDT + cc * 8]) = pa[it];
        *(uint4*)(&Bs[cur ^ 1][row * LDT + cc * 8]) = pb[it];
      }
      __syncthreads();
    }
#endif
    cur ^= 1;
  }

  // epilogue: C/D layout M = r + 8*half, N = mr within each 16x16 tile
#pragma unroll
  for (int i = 0; i < 4; ++i) {
#pragma unroll
    for (int j = 0; j < 2; ++j) {
#pragma unroll
      for (int r = 0; r < 8; ++r) {
        const int m   = bm0 + wr * 64 + i * 16 + r + 8 * half;
        const int col = bn0 + wc * 32 + j * 16 + mr;
        float val = acc[i][j][r] + bias[col];
        if constexpr (EPI == EPI_RES_F32) {
          outF[(size_t)m * N + col] = val + res[(size_t)m * N + col];
        } else if constexpr (EPI == EPI_GELU_BF16) {
          float gl = 0.5f * val * (1.0f + erff(val * 0.70710678118654752f));
          outH[(size_t)m * N + col] = f2bf(gl);
        } else {  // EPI_QKV: bias + phi + mask, scatter to attention layouts
          const int bb = m >> 12, nn = m & 4095;          // N = 4096
          const int t = col >> 10, rem = col & 1023;      // C = 1024
          const int hh = rem >> 6, dd = rem & 63;         // D = 64
          const bool valid = (mask[bb * Nn + nn] == 0);   // mask true = PAD
          if (t == 0) {
            float ph = val > 0.f ? val + 1.f : __expf(val);
            qf[(size_t)((bb * Hh + hh) * Nn + nn) * Dd + dd] = f2bf(ph);
          } else if (t == 1) {
            float ph = val > 0.f ? val + 1.f : __expf(val);
            kf[(size_t)((bb * Hh + hh) * Dd + dd) * Nn + nn] = f2bf(valid ? ph : 0.f);
          } else {
            vv[(size_t)((bb * Hh + hh) * Dd + dd) * Nn + nn] = f2bf(valid ? val : 0.f);
          }
        }
      }
    }
  }
}

// ---------------------------------------------------------------------------
// kv[b,h] = kf^T @ v  (64x64, K = N = 4096) and z[b,h] = sum_n kf.
// kf/v stored [BH, D, N] so BOTH WMMA operands read contiguous 16B chunks
// straight from global (L2-resident, 1 MB per block).  Cross-wave reduction
// via ds_add_f32 atomics in LDS.
// ---------------------------------------------------------------------------
__global__ __launch_bounds__(256)
void kv_kernel(const unsigned short* __restrict__ Kf,  // [BH, D, N]
               const unsigned short* __restrict__ Vv,  // [BH, D, N]
               float* __restrict__ KV,                 // [BH, D, D]
               float* __restrict__ Z) {                // [BH, D]
  __shared__ float accs[64 * 64];
  __shared__ float zsh[4 * 64];
  const int bh = blockIdx.x;
  const int tid = threadIdx.x, lane = tid & 31, wave = tid >> 5;
  const int half = lane >> 4, mr = lane & 15;
  const size_t base = (size_t)bh * Dd * Nn;

  for (int i = tid; i < 4096; i += 256) accs[i] = 0.f;
  __syncthreads();

  const v8f zero8 = {0.f, 0.f, 0.f, 0.f, 0.f, 0.f, 0.f, 0.f};
  v8f acc[4][4];
#pragma unroll
  for (int i = 0; i < 4; ++i)
#pragma unroll
    for (int j = 0; j < 4; ++j) acc[i][j] = zero8;

  for (int c = 0; c < 16; ++c) {            // each wave reduces 512 positions
    const int kbase = wave * 512 + c * 32;
    FragU a[4], b[4];
#pragma unroll
    for (int i = 0; i < 4; ++i) {
      const unsigned short* p = Kf + base + (size_t)(i * 16 + mr) * Nn + kbase + 8 * half;
      a[i].u[0] = *(const uint4*)p;
      a[i].u[1] = *(const uint4*)(p + 16);
    }
#pragma unroll
    for (int j = 0; j < 4; ++j) {
      const unsigned short* p = Vv + base + (size_t)(j * 16 + mr) * Nn + kbase + 8 * half;
      b[j].u[0] = *(const uint4*)p;
      b[j].u[1] = *(const uint4*)(p + 16);
    }
#pragma unroll
    for (int i = 0; i < 4; ++i)
#pragma unroll
      for (int j = 0; j < 4; ++j)
        acc[i][j] = __builtin_amdgcn_wmma_f32_16x16x32_bf16(
            false, a[i].v, false, b[j].v, (short)0, acc[i][j], false, false);
  }

#pragma unroll
  for (int i = 0; i < 4; ++i)
#pragma unroll
    for (int j = 0; j < 4; ++j)
#pragma unroll
      for (int r = 0; r < 8; ++r) {
        int ml = i * 16 + r + 8 * half;
        int nl = j * 16 + mr;
        atomicAdd(&accs[ml * 64 + nl], acc[i][j][r]);   // ds_add_f32
      }

  {  // z partials: thread (d, quarter) sums 1024 contiguous bf16
    const int dd = tid & 63, part = tid >> 6;
    const uint4* p = (const uint4*)(Kf + base + (size_t)dd * Nn + part * 1024);
    float s = 0.f;
    for (int i = 0; i < 128; ++i) {
      uint4 q = p[i];
      s += bf2f((unsigned short)q.x) + bf2f((unsigned short)(q.x >> 16));
      s += bf2f((unsigned short)q.y) + bf2f((unsigned short)(q.y >> 16));
      s += bf2f((unsigned short)q.z) + bf2f((unsigned short)(q.z >> 16));
      s += bf2f((unsigned short)q.w) + bf2f((unsigned short)(q.w >> 16));
    }
    zsh[part * 64 + dd] = s;
  }
  __syncthreads();

  for (int i = tid; i < 4096; i += 256) KV[(size_t)bh * 4096 + i] = accs[i];
  if (tid < 64)
    Z[bh * 64 + tid] = zsh[tid] + zsh[64 + tid] + zsh[128 + tid] + zsh[192 + tid];
}

// ---------------------------------------------------------------------------
// y = (qf @ kv) / max(qf . z, eps); writes Y bf16 in [B*N, C] (heads merged).
// One block per (b,h, 128-row slab).  kv staged transposed bf16 in LDS.
// ---------------------------------------------------------------------------
__global__ __launch_bounds__(256)
void attn_y_kernel(const unsigned short* __restrict__ Qf,  // [BH, N, D]
                   const float* __restrict__ KV,           // [BH, D, D]
                   const float* __restrict__ Z,            // [BH, D]
                   unsigned short* __restrict__ Y) {       // [B*N, C]
  __shared__ __align__(16) unsigned short Bt[64 * 72];  // kv^T: [e][d]
  __shared__ float zshf[64];
  __shared__ float densh[128];
  const int bh = blockIdx.x, nb = blockIdx.y;
  const int b = bh >> 4, h = bh & 15;
  const int n0 = nb * 128;
  const int tid = threadIdx.x, lane = tid & 31, wave = tid >> 5;
  const int half = lane >> 4, mr = lane & 15;
  const size_t qbase = ((size_t)bh * Nn + n0) * Dd;

  for (int i = tid; i < 4096; i += 256) {
    int d = i >> 6, e = i & 63;
    Bt[e * 72 + d] = f2bf(KV[(size_t)bh * 4096 + i]);
  }
  if (tid < 64) zshf[tid] = Z[bh * 64 + tid];
  __syncthreads();

  if (tid < 128) {  // per-row denominator qf . z
    const uint4* p = (const uint4*)(Qf + qbase + (size_t)tid * Dd);
    float s = 0.f;
#pragma unroll
    for (int i = 0; i < 8; ++i) {
      uint4 q = p[i];
      const int d0 = i * 8;
      s += bf2f((unsigned short)q.x) * zshf[d0 + 0] + bf2f((unsigned short)(q.x >> 16)) * zshf[d0 + 1];
      s += bf2f((unsigned short)q.y) * zshf[d0 + 2] + bf2f((unsigned short)(q.y >> 16)) * zshf[d0 + 3];
      s += bf2f((unsigned short)q.z) * zshf[d0 + 4] + bf2f((unsigned short)(q.z >> 16)) * zshf[d0 + 5];
      s += bf2f((unsigned short)q.w) * zshf[d0 + 6] + bf2f((unsigned short)(q.w >> 16)) * zshf[d0 + 7];
    }
    densh[tid] = fmaxf(s, 1e-6f);
  }
  __syncthreads();

  const v8f zero8 = {0.f, 0.f, 0.f, 0.f, 0.f, 0.f, 0.f, 0.f};
  v8f acc[4] = {zero8, zero8, zero8, zero8};
#pragma unroll
  for (int kk = 0; kk < 64; kk += 32) {
    FragU a, bq[4];
    const unsigned short* ap = Qf + qbase + (size_t)(wave * 16 + mr) * Dd + kk + 8 * half;
    a.u[0] = *(const uint4*)ap;
    a.u[1] = *(const uint4*)(ap + 16);
#pragma unroll
    for (int j = 0; j < 4; ++j) {
      const unsigned short* p = Bt + (j * 16 + mr) * 72 + kk + 8 * half;
      bq[j].u[0] = *(const uint4*)p;
      bq[j].u[1] = *(const uint4*)(p + 16);
    }
#pragma unroll
    for (int j = 0; j < 4; ++j)
      acc[j] = __builtin_amdgcn_wmma_f32_16x16x32_bf16(
          false, a.v, false, bq[j].v, (short)0, acc[j], false, false);
  }

#pragma unroll
  for (int j = 0; j < 4; ++j)
#pragma unroll
    for (int r = 0; r < 8; ++r) {
      const int ml = wave * 16 + r + 8 * half;
      const int n = n0 + ml;
      const int e = j * 16 + mr;
      float yv = acc[j][r] / densh[ml];
      Y[((size_t)b * Nn + n) * Cc + h * Dd + e] = f2bf(yv);
    }
}

// ---------------------------------------------------------------------------
// launch
// ---------------------------------------------------------------------------
extern "C" void kernel_launch(void* const* d_in, const int* in_sizes, int n_in,
                              void* d_out, int out_size, void* d_ws, size_t ws_size,
                              hipStream_t stream) {
  (void)in_sizes; (void)n_in; (void)out_size; (void)ws_size;
  const float* src   = (const float*)d_in[0];
  const unsigned char* mask = (const unsigned char*)d_in[1];  // bool (1 byte)
  const float* qkv_w = (const float*)d_in[2];
  const float* qkv_b = (const float*)d_in[3];
  const float* out_w = (const float*)d_in[4];
  const float* out_b = (const float*)d_in[5];
  const float* w1    = (const float*)d_in[6];
  const float* b1    = (const float*)d_in[7];
  const float* w2    = (const float*)d_in[8];
  const float* b2    = (const float*)d_in[9];
  const float* ln1_g = (const float*)d_in[10];
  const float* ln1_b = (const float*)d_in[11];
  const float* ln2_g = (const float*)d_in[12];
  const float* ln2_b = (const float*)d_in[13];
  float* out = (float*)d_out;

  char* ws = (char*)d_ws;
  auto alloc = [&](size_t bytes) -> char* {
    char* p = ws;
    ws += (bytes + 255) & ~(size_t)255;
    return p;
  };
  unsigned short* WqkvH = (unsigned short*)alloc((size_t)3 * Cc * Cc * 2);
  unsigned short* WoutH = (unsigned short*)alloc((size_t)Cc * Cc * 2);
  unsigned short* W1H   = (unsigned short*)alloc((size_t)FFd * Cc * 2);
  unsigned short* W2H   = (unsigned short*)alloc((size_t)Cc * FFd * 2);
  unsigned short* X1H   = (unsigned short*)alloc((size_t)Mrows * Cc * 2);
  unsigned short* QfH   = (unsigned short*)alloc((size_t)Mrows * Cc * 2);
  unsigned short* KfH   = (unsigned short*)alloc((size_t)Mrows * Cc * 2);
  unsigned short* VvH   = (unsigned short*)alloc((size_t)Mrows * Cc * 2);
  float*          KVF   = (float*)alloc((size_t)Bb * Hh * Dd * Dd * 4);
  float*          ZF    = (float*)alloc((size_t)Bb * Hh * Dd * 4);
  unsigned short* YH    = (unsigned short*)alloc((size_t)Mrows * Cc * 2);
  float*          XresF = (float*)alloc((size_t)Mrows * Cc * 4);
  unsigned short* X2H   = (unsigned short*)alloc((size_t)Mrows * Cc * 2);
  unsigned short* HbH   = (unsigned short*)alloc((size_t)Mrows * FFd * 2);

  // weights -> bf16
  f32_to_bf16_kernel<<<1024, 256, 0, stream>>>(qkv_w, WqkvH, 3 * Cc * Cc);
  f32_to_bf16_kernel<<<1024, 256, 0, stream>>>(out_w, WoutH, Cc * Cc);
  f32_to_bf16_kernel<<<1024, 256, 0, stream>>>(w1, W1H, FFd * Cc);
  f32_to_bf16_kernel<<<1024, 256, 0, stream>>>(w2, W2H, Cc * FFd);

  // LN1
  layernorm_bf16_kernel<<<Mrows, 256, 0, stream>>>(src, ln1_g, ln1_b, X1H);

  // QKV GEMM + bias + phi + mask scatter
  gemm_bf16_kernel<EPI_QKV><<<dim3(3 * Cc / 128, Mrows / 128), 256, 0, stream>>>(
      X1H, WqkvH, qkv_b, nullptr, nullptr, nullptr, QfH, KfH, VvH, mask,
      Mrows, 3 * Cc, Cc);

  // kv = kf^T @ v ; z = sum kf
  kv_kernel<<<Bb * Hh, 256, 0, stream>>>(KfH, VvH, KVF, ZF);

  // y = qf @ kv / (qf . z)
  attn_y_kernel<<<dim3(Bb * Hh, Nn / 128), 256, 0, stream>>>(QfH, KVF, ZF, YH);

  // out projection + residual -> x
  gemm_bf16_kernel<EPI_RES_F32><<<dim3(Cc / 128, Mrows / 128), 256, 0, stream>>>(
      YH, WoutH, out_b, src, XresF, nullptr, nullptr, nullptr, nullptr, nullptr,
      Mrows, Cc, Cc);

  // LN2
  layernorm_bf16_kernel<<<Mrows, 256, 0, stream>>>(XresF, ln2_g, ln2_b, X2H);

  // FFN1 + exact GeLU
  gemm_bf16_kernel<EPI_GELU_BF16><<<dim3(FFd / 128, Mrows / 128), 256, 0, stream>>>(
      X2H, W1H, b1, nullptr, nullptr, HbH, nullptr, nullptr, nullptr, nullptr,
      Mrows, FFd, Cc);

  // FFN2 + residual -> out
  gemm_bf16_kernel<EPI_RES_F32><<<dim3(Cc / 128, Mrows / 128), 256, 0, stream>>>(
      HbH, W2H, b2, XresF, out, nullptr, nullptr, nullptr, nullptr, nullptr,
      Mrows, Cc, FFd);
}
